// MemoryNetwork_59545426591806
// MI455X (gfx1250) — matrix operand order repaired and checked by reading, not compiled, not run on hardware
//
#include <hip/hip_runtime.h>

// ---------------- types ----------------
typedef unsigned short u16;
typedef __attribute__((ext_vector_type(16))) __bf16 v16bf;
typedef __attribute__((ext_vector_type(8)))  __bf16 v8bf;
typedef __attribute__((ext_vector_type(8)))  float  v8f;
typedef __attribute__((ext_vector_type(4)))  int    v4i;

// ---------------- problem constants ----------------
#define BB   32768      // batch
#define KK   768        // input dim
#define EE   256        // embedding dim
#define DD   9          // domains
#define MM   10         // memory slots
#define NW   512        // combined weight rows (topic 0..255, domain 256..511)
#define NP   96         // padded D*M (90 -> 96)
#define NKC  (KK / 32)  // 24 K-chunks
#define TAU  32.0f

// padded LDS row strides (u16 elements) -> conflict-free b128 fragment loads
#define BROW 40         // 80 B   (16 lanes hit disjoint bank quads)
#define AROW 40
#define ROWP 264        // 528 B  (132 dwords; lanes hit banks 0,4,8,...,60)

// workspace layout (bytes)
//  Wcat stored K-chunk tiled: [NKC][512][32] bf16 -> each K-slice is one
//  contiguous 32KB block, so async-to-LDS requests are fully coalesced.
#define WS_WCAT_OFF 0                          // bf16 [24][512][32]
#define WS_MEM_OFF  (NW * KK * 2)              // bf16 [96][256]

// dynamic LDS carve (bytes)
//  phase 1:   sB[2][512*40]u16 @0 (81920) | sA[2][32*40]u16 @81920 (5120)   -> ends 87040
//  phase 2/3: sMem[96*264]u16 @0 (50688) | sTD[2][32*264]u16 @50688 (33792)
//             sXY[2][32][96]f32 @84480 (24576)                              -> ends 109056
//  always:    sSS[32]f32 @109056 (128)
//  phase 4:   sDA[32*9]f32 @0
#define SM_B     0
#define SM_A     81920
#define SM_MEM   0
#define SM_TD    50688
#define SM_XY    84480
#define SM_SS    109056
#define SM_TOTAL 109184

// ---------------- CDNA5 async global->LDS path ----------------
#if defined(__HIP_DEVICE_COMPILE__) && defined(__has_builtin)
#  if __has_builtin(__builtin_amdgcn_global_load_async_to_lds_b128)
#    define HAVE_ASYNC_LDS 1
#  endif
#endif
#ifndef HAVE_ASYNC_LDS
#  define HAVE_ASYNC_LDS 0
#endif

__device__ __forceinline__ void copy16(const u16* g, u16* l) {
#if HAVE_ASYNC_LDS
    __builtin_amdgcn_global_load_async_to_lds_b128(
        (__attribute__((address_space(1))) v4i*)(unsigned long long)(g),
        (__attribute__((address_space(3))) v4i*)(unsigned int)(unsigned long long)(l),
        0, 0);
#else
    *(uint4*)l = *(const uint4*)g;
#endif
}

__device__ __forceinline__ void wait_async_lds() {
#if HAVE_ASYNC_LDS
    asm volatile("s_wait_asynccnt 0x0" ::: "memory");
#endif
}

// ---------------- helpers ----------------
__device__ __forceinline__ u16 f2bf(float f) {
    union { float f; unsigned u; } c; c.f = f;
    unsigned b = c.u;
    b = b + 0x7FFFu + ((b >> 16) & 1u);      // round-to-nearest-even
    return (u16)(b >> 16);
}

// A-matrix 16-bit fragment (16x32): lanes 0-15 hold K 0..15,
// lanes 16-31 hold K 16..31 for the same rows (matches frag_b order).
__device__ __forceinline__ v16bf frag_a(const u16* rowp, int hs) {
    v8bf lo = *(const v8bf*)(rowp + hs * 16);
    v8bf hi = *(const v8bf*)(rowp + hs * 16 + 8);
    v16bf a;
#pragma unroll
    for (int i = 0; i < 8; ++i) { a[i] = lo[i]; a[i + 8] = hi[i]; }
    return a;
}

// B-matrix 16-bit fragment (32x16): lanes 0-15 hold K=0..15, lanes 16-31 K=16..31
// for column = lane%16 (contiguous K halves, same order as frag_a).
__device__ __forceinline__ v16bf frag_b(const u16* colp, int hs) {
    v8bf lo = *(const v8bf*)(colp + hs * 16);
    v8bf hi = *(const v8bf*)(colp + hs * 16 + 8);
    v16bf b;
#pragma unroll
    for (int i = 0; i < 8; ++i) { b[i] = lo[i]; b[i + 8] = hi[i]; }
    return b;
}

__device__ __forceinline__ v8f wmma_bf(v16bf a, v16bf b, v8f c) {
    return __builtin_amdgcn_wmma_f32_16x16x32_bf16(
        false, a, false, b, (short)0, c, false, false);
}

// ---------------- prep kernels ----------------
// Writes Wcat K-chunk tiled: dst[kc][n][k'] with kc = k/32, k' = k%32.
__global__ void prep_wcat(const float* __restrict__ Wt,
                          const float* __restrict__ Wd,
                          u16* __restrict__ Wcat) {
    int idx = blockIdx.x * 256 + threadIdx.x;
    if (idx >= NW * KK) return;
    int n = idx / KK, k = idx - n * KK;
    float v = (n < EE) ? Wt[n * KK + k] : Wd[(n - EE) * KK + k];
    Wcat[(k >> 5) * (NW * 32) + n * 32 + (k & 31)] = f2bf(v);
}

__global__ void prep_mem(const float* __restrict__ mem, u16* __restrict__ memB) {
    int idx = blockIdx.x * 256 + threadIdx.x;
    if (idx >= NP * EE) return;
    int n = idx / EE, e = idx - n * EE;
    float v = (n < DD * MM) ? mem[n * EE + e] : 0.0f;   // memory is [9][10][256] = [90][256]
    memB[idx] = f2bf(v);
}

// ---------------- fused kernel ----------------
// block: 256 threads = 8 waves (wave32). Waves: wm = wid&1 (M), wn = wid>>1 (N).
// block tile: 32 rows x 512 cols; wave tile: 16 x 128 = 8 WMMA accum tiles.
// Double-buffered K-loop: async-DMA B tile + register-prefetched A tile.
__global__ __launch_bounds__(256)
void memnet_fused(const float* __restrict__ feature,
                  const u16* __restrict__ Wcat,
                  const u16* __restrict__ memB,
                  float* __restrict__ out) {
    extern __shared__ char smem[];
    u16*   sB   = (u16*)(smem + SM_B);     // [2][512][BROW] bf16
    u16*   sA   = (u16*)(smem + SM_A);     // [2][32][AROW]  bf16
    u16*   sMem = (u16*)(smem + SM_MEM);   // [96][ROWP] bf16
    u16*   sTD  = (u16*)(smem + SM_TD);    // [2][32][ROWP] bf16 (topic, dom)
    float* sXY  = (float*)(smem + SM_XY);  // [2][32][96]
    float* sSS  = (float*)(smem + SM_SS);  // [32] sum of squares (non-overlapping)
    float* sDA  = (float*)(smem);          // [32][9]

    const int tid  = threadIdx.x;
    const int wid  = tid >> 5;
    const int lane = tid & 31;
    const int lr   = lane & 15;
    const int hs   = lane >> 4;
    const int wm   = wid & 1;
    const int wn   = wid >> 1;
    const int row0 = blockIdx.x * 32;

    v8f acc[8] = {};

    if (tid < 32) sSS[tid] = 0.0f;

    // stage B tile for K-chunk kc into buffer buf (8 x b128 per thread).
    // Wcat is K-chunk tiled, so consecutive lanes fetch consecutive 16B chunks.
    auto stageB = [&](int kc, int buf) {
        u16* dst = sB + buf * (NW * BROW);
        const u16* src = Wcat + kc * (NW * 32);
#pragma unroll
        for (int i = 0; i < 8; ++i) {
            int c   = tid + i * 256;     // chunk id in [0, 2048)
            int n   = c >> 2;            // weight row
            int seg = c & 3;             // 8-halfword segment
            copy16(src + c * 8, dst + n * BROW + seg * 8);
        }
    };

    const int ar = tid >> 3;            // A row this thread stages (0..31)
    const int ak = (tid & 7) << 2;      // A k-offset within chunk (0..28)
    const float* arow = feature + (row0 + ar) * KK + ak;

    // -------- pipeline prologue --------
    stageB(0, 0);
    float4 fv = *(const float4*)(arow);

    // ---------------- phase 1: C[32x512] = feature_raw @ Wcat^T ----------------
    for (int kc = 0; kc < NKC; ++kc) {
        const int cur = kc & 1;
        // commit prefetched A tile (f32 -> bf16) + accumulate row sum-of-squares
        u16* sa = sA + cur * (32 * AROW);
        sa[ar * AROW + ak + 0] = f2bf(fv.x);
        sa[ar * AROW + ak + 1] = f2bf(fv.y);
        sa[ar * AROW + ak + 2] = f2bf(fv.z);
        sa[ar * AROW + ak + 3] = f2bf(fv.w);
        atomicAdd(&sSS[ar], fv.x * fv.x + fv.y * fv.y + fv.z * fv.z + fv.w * fv.w);

        wait_async_lds();          // B tile for `cur` landed in LDS
        __syncthreads();           // A stores + everyone's async B visible

        if (kc + 1 < NKC) {        // kick off next chunk while we compute
            stageB(kc + 1, cur ^ 1);
            fv = *(const float4*)(arow + (kc + 1) * 32);
        }

        // compute: batch fragment loads, then burst 8 WMMAs
        const u16* sbc = sB + cur * (NW * BROW);
        v16bf af = frag_a(sA + cur * (32 * AROW) + (wm * 16 + lr) * AROW, hs);
        v16bf bfr[8];
#pragma unroll
        for (int t = 0; t < 8; ++t)
            bfr[t] = frag_b(sbc + (wn * 128 + t * 16 + lr) * BROW, hs);
#pragma unroll
        for (int t = 0; t < 8; ++t)
            acc[t] = wmma_bf(af, bfr[t], acc[t]);
    }
    __syncthreads();

    // ---------------- phase 2a: normalize + store topic/dom tiles as bf16 ----------------
#pragma unroll
    for (int i = 0; i < 8; ++i) {
        int rl = wm * 16 + hs * 8 + i;
        float rs = rsqrtf(sSS[rl]);
#pragma unroll
        for (int t = 0; t < 8; ++t) {
            int col = wn * 128 + t * 16 + lr;
            int src = col >> 8;            // 0 = topic, 1 = dom
            int e   = col & 255;
            sTD[src * (32 * ROWP) + rl * ROWP + e] = f2bf(acc[t][i] * rs);
        }
    }
    __syncthreads();   // sA/sB dead past here

    // ---------------- phase 2b: stage bf16 memory bank [96][256] into padded LDS --------
    for (int j = tid; j < NP * 32; j += 256) {
        int row = j >> 5, seg = j & 31;
        copy16(memB + row * 256 + seg * 8, sMem + row * ROWP + seg * 8);
    }
    wait_async_lds();
    __syncthreads();

    // ---------------- phase 3: X = topic @ mem^T, Y = dom @ mem^T (K=256, N=96) ----------
    // 24 tiles (2 src x 2 mtiles x 6 ntiles) -> 3 per wave, full EXEC.
#pragma unroll
    for (int it = 0; it < 3; ++it) {
        int idx = wid + it * 8;
        int src = idx / 12;
        int rem = idx % 12;
        int mt  = rem / 6;
        int nt  = rem % 6;
        v8f a2 = {};
#pragma unroll
        for (int kc2 = 0; kc2 < EE / 32; ++kc2) {
            v16bf af  = frag_a(sTD + src * (32 * ROWP) + (mt * 16 + lr) * ROWP + kc2 * 32, hs);
            v16bf bf_ = frag_b(sMem + (nt * 16 + lr) * ROWP + kc2 * 32, hs);
            a2 = wmma_bf(af, bf_, a2);
        }
        int col = nt * 16 + lr;
#pragma unroll
        for (int i = 0; i < 8; ++i) {
            int rl = mt * 16 + hs * 8 + i;
            sXY[src * (32 * NP) + rl * NP + col] = a2[i];
        }
    }
    __syncthreads();

    // ---------------- phase 4a: per-(row, d) softmax over m, combine with y --------------
    for (int p = tid; p < 32 * DD; p += 256) {
        int rl = p / DD, d = p - rl * DD;
        const float* xp = sXY + rl * NP + d * MM;
        const float* yp = sXY + (32 * NP) + rl * NP + d * MM;
        float xs[MM], mx = -3.0e38f;
#pragma unroll
        for (int m = 0; m < MM; ++m) { xs[m] = TAU * xp[m]; mx = fmaxf(mx, xs[m]); }
        float se = 0.0f, da = 0.0f;
#pragma unroll
        for (int m = 0; m < MM; ++m) {
            float e = __expf(xs[m] - mx);
            se += e;
            da += e * yp[m];
        }
        sDA[rl * DD + d] = da / se;
    }
    __syncthreads();

    // ---------------- phase 4b: per-row softmax over d -> output [B][9] ------------------
    if (tid < 32) {
        float v[DD], mx = -3.0e38f;
#pragma unroll
        for (int d = 0; d < DD; ++d) { v[d] = TAU * sDA[tid * DD + d]; mx = fmaxf(mx, v[d]); }
        float se = 0.0f;
#pragma unroll
        for (int d = 0; d < DD; ++d) { v[d] = __expf(v[d] - mx); se += v[d]; }
        float inv = 1.0f / se;
#pragma unroll
        for (int d = 0; d < DD; ++d) out[(row0 + tid) * DD + d] = v[d] * inv;
    }
}

// ---------------- launcher ----------------
extern "C" void kernel_launch(void* const* d_in, const int* in_sizes, int n_in,
                              void* d_out, int out_size, void* d_ws, size_t ws_size,
                              hipStream_t stream) {
    (void)in_sizes; (void)n_in; (void)out_size; (void)ws_size;
    const float* feature = (const float*)d_in[0];
    // d_in[1] = category (unused by the forward math)
    const float* Wt  = (const float*)d_in[2];
    const float* Wd  = (const float*)d_in[3];
    const float* mem = (const float*)d_in[4];
    float* out = (float*)d_out;

    u16* Wcat = (u16*)((char*)d_ws + WS_WCAT_OFF);
    u16* memB = (u16*)((char*)d_ws + WS_MEM_OFF);

    prep_wcat<<<(NW * KK + 255) / 256, 256, 0, stream>>>(Wt, Wd, Wcat);
    prep_mem<<<(NP * EE + 255) / 256, 256, 0, stream>>>(mem, memB);
    memnet_fused<<<BB / 32, 256, SM_TOTAL, stream>>>(feature, Wcat, memB, out);
}
